// GATNet_19018115187323
// MI455X (gfx1250) — matrix operand order repaired and checked by reading, not compiled, not run on hardware
//
#include <hip/hip_runtime.h>
#include <hip/hip_bf16.h>

typedef __attribute__((ext_vector_type(16))) __bf16 v16bf;
typedef __attribute__((ext_vector_type(8)))  float  v8f;

#define F_IN 512
#define HC1 64   // H1*C1
#define H1 8
#define C1 8
#define C2 40
#define NEG_SLOPE 0.2f

// Native fptrunc: gfx1250 has full bf16 VALU support, so let clang emit the
// hardware cvt (packed v_cvt_*bf16* where possible) instead of bit-twiddling.
static __device__ __forceinline__ __bf16 f2bf(float f) {
    return static_cast<__bf16>(f);
}
// order-preserving float -> uint key (for atomicMax-based segment max)
static __device__ __forceinline__ unsigned fkey(float f) {
    unsigned u = __builtin_bit_cast(unsigned, f);
    return (u & 0x80000000u) ? ~u : (u | 0x80000000u);
}
static __device__ __forceinline__ float fdec(unsigned k) {
    unsigned u = (k & 0x80000000u) ? (k ^ 0x80000000u) : ~k;
    return __builtin_bit_cast(float, u);
}
static __device__ __forceinline__ float lrelu(float f) {
    return f > 0.0f ? f : NEG_SLOPE * f;
}

// ---------------- zero init ----------------
__global__ void zero_u32(unsigned* p, long long n) {
    long long i = (long long)blockIdx.x * blockDim.x + threadIdx.x;
    if (i < n) p[i] = 0u;
}

// ------- weight transpose + f32->bf16 (B fragments want [col][K] contiguous) -------
__global__ void prep_weights(const float* W1, const float* W2, __bf16* W1t, __bf16* W2t) {
    int idx = blockIdx.x * blockDim.x + threadIdx.x;
    const int n1 = HC1 * F_IN;           // 64 x 512
    const int n2 = 48 * HC1;             // 48 x 64 (cols 40..47 zero-padded)
    if (idx < n1) {
        int n = idx / F_IN, k = idx % F_IN;
        W1t[idx] = f2bf(W1[(size_t)k * HC1 + n]);
    } else if (idx < n1 + n2) {
        int j = idx - n1;
        int n = j / HC1, k = j % HC1;
        W2t[j] = (n < C2) ? f2bf(W2[(size_t)k * C2 + n]) : f2bf(0.0f);
    }
}

// ---------------- WMMA GEMM: C[nrows,ncols] = A[nrows,K] * Bt^T ----------------
// One wave computes a 16x16 tile; wave w in the block owns N-tile w.
// A converted f32->bf16 in-register; Bt is bf16 [colPad][K] (pre-transposed).
template<int K>
__global__ void gemm_bf16_wmma(const float* __restrict__ A,
                               const __bf16* __restrict__ Bt,
                               float* __restrict__ C,
                               int nrows, int ncols, int ldc) {
    const int lane = threadIdx.x & 31;
    const int wave = threadIdx.x >> 5;       // N-tile index
    const int l16  = lane & 15;              // A row / B,D column within tile
    const int kh   = lane >> 4;              // K-half select
    int grow = blockIdx.x * 16 + l16;
    const float* ar = A + (size_t)(grow < nrows ? grow : nrows - 1) * K;
    const __bf16* br = Bt + (size_t)(wave * 16 + l16) * K;

    v8f acc = {};
#pragma unroll
    for (int kk = 0; kk < K; kk += 32) {
        // A fragment: row l16, contiguous K = kk + kh*16 .. +15
        const float4* p0 = (const float4*)(ar + kk + kh * 16);
        const float4* p1 = (const float4*)(ar + kk + kh * 16 + 8);
        float4 f0 = p0[0], f1 = p0[1], f2 = p1[0], f3 = p1[1];
        v16bf a;
        a[0] = f2bf(f0.x); a[1] = f2bf(f0.y); a[2]  = f2bf(f0.z); a[3]  = f2bf(f0.w);
        a[4] = f2bf(f1.x); a[5] = f2bf(f1.y); a[6]  = f2bf(f1.z); a[7]  = f2bf(f1.w);
        a[8] = f2bf(f2.x); a[9] = f2bf(f2.y); a[10] = f2bf(f2.z); a[11] = f2bf(f2.w);
        a[12] = f2bf(f3.x); a[13] = f2bf(f3.y); a[14] = f2bf(f3.z); a[15] = f2bf(f3.w);
        // B fragment: column l16, contiguous K = kk + kh*16 .. +15 (32B load)
        v16bf b = *(const v16bf*)(br + kk + kh * 16);
        acc = __builtin_amdgcn_wmma_f32_16x16x32_bf16(
            false, a, false, b, (short)0, acc, false, false);
    }
    int ncol = wave * 16 + l16;
#pragma unroll
    for (int r = 0; r < 8; ++r) {
        int m = blockIdx.x * 16 + r + 8 * kh;      // D: lane L, vgpr r -> M=r+8*(L/16)
        if (m < nrows && ncol < ncols)
            C[(size_t)m * ldc + ncol] = acc[r];
    }
}

// ---------------- attention logits: al[n,h] = <feat[n,h,:], a[h,:]> ----------------
__global__ void attn_logits(const float* __restrict__ feat,
                            const float* __restrict__ a_src,
                            const float* __restrict__ a_dst,
                            float* __restrict__ als, float* __restrict__ ald,
                            int Nn, int H, int Cc) {
    int idx = blockIdx.x * blockDim.x + threadIdx.x;
    if (idx >= Nn * H) return;
    int n = idx / H, h = idx % H;
    const float* f = feat + (size_t)n * H * Cc + h * Cc;
    float s = 0.f, d = 0.f;
    for (int c = 0; c < Cc; ++c) { float v = f[c]; s += v * a_src[h * Cc + c]; d += v * a_dst[h * Cc + c]; }
    als[idx] = s; ald[idx] = d;
}

// ---------------- edge pass 1: segment max via ordered-key atomicMax ----------------
__global__ void edge_max(const int* __restrict__ ei, long long E, int Nn,
                         const float* __restrict__ als, const float* __restrict__ ald,
                         unsigned* __restrict__ mkey, int H) {
    long long e = (long long)blockIdx.x * blockDim.x + threadIdx.x;
    if (e >= E + Nn) return;
    int s = (e < E) ? ei[e] : (int)(e - E);
    int d = (e < E) ? ei[E + e] : (int)(e - E);
    for (int h = 0; h < H; ++h) {
        float v = lrelu(als[(size_t)s * H + h] + ald[(size_t)d * H + h]);
        atomicMax(&mkey[(size_t)d * H + h], fkey(v));
    }
}

// ---------------- edge pass 2: denom = segment_sum(exp(e - m)) ----------------
__global__ void edge_sumexp(const int* __restrict__ ei, long long E, int Nn,
                            const float* __restrict__ als, const float* __restrict__ ald,
                            const unsigned* __restrict__ mkey, float* __restrict__ denom, int H) {
    long long e = (long long)blockIdx.x * blockDim.x + threadIdx.x;
    if (e >= E + Nn) return;
    int s = (e < E) ? ei[e] : (int)(e - E);
    int d = (e < E) ? ei[E + e] : (int)(e - E);
    for (int h = 0; h < H; ++h) {
        float v = lrelu(als[(size_t)s * H + h] + ald[(size_t)d * H + h]);
        float ex = __expf(v - fdec(mkey[(size_t)d * H + h]));
        atomicAdd(&denom[(size_t)d * H + h], ex);
    }
}

// ---------------- edge pass 3: agg[dst] += alpha * feat[src], 8-channel chunks ----------------
__global__ void edge_agg(const int* __restrict__ ei, long long E, int Nn,
                         const float* __restrict__ als, const float* __restrict__ ald,
                         const unsigned* __restrict__ mkey, const float* __restrict__ denom,
                         const float* __restrict__ feat, float* __restrict__ agg,
                         int H, int Cc) {
    int nchunk = (H * Cc) / 8;
    long long idx = (long long)blockIdx.x * blockDim.x + threadIdx.x;
    long long e = idx / nchunk;
    if (e >= E + Nn) return;
    int j = (int)(idx % nchunk);
    int s = (e < E) ? ei[e] : (int)(e - E);
    int d = (e < E) ? ei[E + e] : (int)(e - E);
    int h  = (j * 8) / Cc;
    int cb = (j * 8) % Cc;
    const float* fs = feat + (size_t)s * H * Cc + h * Cc + cb;
    // Kick off the random gather early (global_prefetch_b8; no LOADcnt cost)
    // so it overlaps the dependent softmax scalar chain below.
    __builtin_prefetch(fs, 0, 3);
    float v  = lrelu(als[(size_t)s * H + h] + ald[(size_t)d * H + h]);
    float ex = __expf(v - fdec(mkey[(size_t)d * H + h]));
    float coef = ex / (denom[(size_t)d * H + h] + 1e-16f);
    float* od = agg + (size_t)d * H * Cc + h * Cc + cb;
#pragma unroll
    for (int c = 0; c < 8; ++c)
        atomicAdd(od + c, coef * fs[c]);
}

// ---------------- bias + relu (between layers) ----------------
__global__ void bias_relu(float* __restrict__ a, const float* __restrict__ b, long long n, int cols) {
    long long i = (long long)blockIdx.x * blockDim.x + threadIdx.x;
    if (i >= n) return;
    float v = a[i] + b[i % cols];
    a[i] = v > 0.f ? v : 0.f;
}

// ---------------- final: bias + log_softmax over 40 classes ----------------
__global__ void logsoftmax_out(const float* __restrict__ a, const float* __restrict__ b,
                               float* __restrict__ out, int Nn) {
    int n = blockIdx.x * blockDim.x + threadIdx.x;
    if (n >= Nn) return;
    float z[C2];
    float mx = -3.0e38f;
    for (int c = 0; c < C2; ++c) { z[c] = a[(size_t)n * C2 + c] + b[c]; mx = fmaxf(mx, z[c]); }
    float s = 0.f;
    for (int c = 0; c < C2; ++c) s += __expf(z[c] - mx);
    float lse = mx + __logf(s);
    for (int c = 0; c < C2; ++c) out[(size_t)n * C2 + c] = z[c] - lse;
}

extern "C" void kernel_launch(void* const* d_in, const int* in_sizes, int n_in,
                              void* d_out, int out_size, void* d_ws, size_t ws_size,
                              hipStream_t stream) {
    const float* x      = (const float*)d_in[0];
    const int*   ei     = (const int*)d_in[1];
    const float* W1     = (const float*)d_in[2];
    const float* a_src1 = (const float*)d_in[3];
    const float* a_dst1 = (const float*)d_in[4];
    const float* b1     = (const float*)d_in[5];
    const float* W2     = (const float*)d_in[6];
    const float* a_src2 = (const float*)d_in[7];
    const float* a_dst2 = (const float*)d_in[8];
    const float* b2     = (const float*)d_in[9];
    float* out = (float*)d_out;

    const int  Nn = in_sizes[0] / F_IN;          // 100000
    const long long E = in_sizes[1] / 2;         // 3.2M
    const long long EP = E + Nn;                 // with self loops

    // -------- workspace layout (256B aligned) --------
    char* ws = (char*)d_ws;
    size_t off = 0;
    auto alloc = [&](size_t bytes) { size_t r = off; off = (off + bytes + 255) & ~(size_t)255; return r; };
    float*    h1     = (float*)   (ws + alloc((size_t)Nn * HC1 * 4));
    float*    als1   = (float*)   (ws + alloc((size_t)Nn * H1 * 4));
    float*    ald1   = (float*)   (ws + alloc((size_t)Nn * H1 * 4));
    unsigned* mkey1  = (unsigned*)(ws + alloc((size_t)Nn * H1 * 4));
    float*    denom1 = (float*)   (ws + alloc((size_t)Nn * H1 * 4));
    float*    agg1   = (float*)   (ws + alloc((size_t)Nn * HC1 * 4));
    float*    h2     = (float*)   (ws + alloc((size_t)Nn * C2 * 4));
    float*    als2   = (float*)   (ws + alloc((size_t)Nn * 4));
    float*    ald2   = (float*)   (ws + alloc((size_t)Nn * 4));
    unsigned* mkey2  = (unsigned*)(ws + alloc((size_t)Nn * 4));
    float*    denom2 = (float*)   (ws + alloc((size_t)Nn * 4));
    float*    agg2   = (float*)   (ws + alloc((size_t)Nn * C2 * 4));
    __bf16*   W1t    = (__bf16*)  (ws + alloc((size_t)HC1 * F_IN * 2));
    __bf16*   W2t    = (__bf16*)  (ws + alloc((size_t)48 * HC1 * 2));

    const int mtiles = (Nn + 15) / 16;
    auto blks = [](long long n, int b) { return (unsigned)((n + b - 1) / b); };

    // 0) weights -> transposed bf16
    prep_weights<<<blks(HC1 * F_IN + 48 * HC1, 256), 256, 0, stream>>>(W1, W2, W1t, W2t);

    // 1) zero accumulators (deterministic under graph replay)
    zero_u32<<<blks((long long)Nn * H1, 256), 256, 0, stream>>>(mkey1, (long long)Nn * H1);
    zero_u32<<<blks((long long)Nn * H1, 256), 256, 0, stream>>>((unsigned*)denom1, (long long)Nn * H1);
    zero_u32<<<blks((long long)Nn * HC1, 256), 256, 0, stream>>>((unsigned*)agg1, (long long)Nn * HC1);
    zero_u32<<<blks(Nn, 256), 256, 0, stream>>>(mkey2, Nn);
    zero_u32<<<blks(Nn, 256), 256, 0, stream>>>((unsigned*)denom2, Nn);
    zero_u32<<<blks((long long)Nn * C2, 256), 256, 0, stream>>>((unsigned*)agg2, (long long)Nn * C2);

    // ---- layer 1 ----
    // h1 = x @ W1  (bf16 WMMA, 4 N-tiles -> 4 waves/block)
    gemm_bf16_wmma<F_IN><<<mtiles, 128, 0, stream>>>(x, W1t, h1, Nn, HC1, HC1);
    attn_logits<<<blks((long long)Nn * H1, 256), 256, 0, stream>>>(h1, a_src1, a_dst1, als1, ald1, Nn, H1, C1);
    edge_max   <<<blks(EP, 256), 256, 0, stream>>>(ei, E, Nn, als1, ald1, mkey1, H1);
    edge_sumexp<<<blks(EP, 256), 256, 0, stream>>>(ei, E, Nn, als1, ald1, mkey1, denom1, H1);
    edge_agg   <<<blks(EP * (H1 * C1 / 8), 256), 256, 0, stream>>>(ei, E, Nn, als1, ald1, mkey1, denom1, h1, agg1, H1, C1);
    bias_relu  <<<blks((long long)Nn * HC1, 256), 256, 0, stream>>>(agg1, b1, (long long)Nn * HC1, HC1);

    // ---- layer 2 ----
    // h2 = relu_h @ W2  (K=64, 3 N-tiles -> 3 waves/block, cols 40..47 padded)
    gemm_bf16_wmma<HC1><<<mtiles, 96, 0, stream>>>(agg1, W2t, h2, Nn, C2, C2);
    attn_logits<<<blks(Nn, 256), 256, 0, stream>>>(h2, a_src2, a_dst2, als2, ald2, Nn, 1, C2);
    edge_max   <<<blks(EP, 256), 256, 0, stream>>>(ei, E, Nn, als2, ald2, mkey2, 1);
    edge_sumexp<<<blks(EP, 256), 256, 0, stream>>>(ei, E, Nn, als2, ald2, mkey2, denom2, 1);
    edge_agg   <<<blks(EP * (C2 / 8), 256), 256, 0, stream>>>(ei, E, Nn, als2, ald2, mkey2, denom2, h2, agg2, 1, C2);

    // ---- output: bias + log_softmax ----
    logsoftmax_out<<<blks(Nn, 256), 256, 0, stream>>>(agg2, b2, out, Nn);
}
